// MonotonicJoinerV2_88450556494795
// MI455X (gfx1250) — matrix operand order repaired and checked by reading, not compiled, not run on hardware
//
#include <hip/hip_runtime.h>
#include <hip/hip_bf16.h>
#include <stdint.h>

// ---------------------------------------------------------------------------
// MonotonicJoinerV2 for MI455X (gfx1250, wave32, WMMA bf16)
//   enc_proj = src @ W1[:512]   (WMMA bf16, f32 accum)
//   pred_proj = tgt @ W1[512:]  (WMMA bf16, f32 accum)
//   out = relu(enc_proj[b,t] + pred_proj[b,u] + b1) @ W2 + b2   (fused, LDS h)
// Round 3: scheduling fence forces the whole per-k-step load clause ahead of
// the WMMA burst (progressive s_wait_loadcnt, loads pipeline across k-steps);
// WGP-scope global_prefetch_b8 (inline asm) warms WGP$ for the next k-step.
// ---------------------------------------------------------------------------

typedef __attribute__((ext_vector_type(16))) __bf16 v16bf;
typedef __attribute__((ext_vector_type(8)))  float  v8f;

#define E_DIM 512
#define HID   640
#define OUT_D 1024
#define B_SZ  8
#define T_SZ  256
#define U_SZ  64
#define HROW_B 1296   // LDS h row stride: (640+8) bf16 = 16B aligned, bank-staggered

union BFragU { uint4 q[2]; v16bf v; };
union PackU  { __bf16 b[4]; uint2 u; };

__device__ __forceinline__ float reluf(float x) { return x > 0.f ? x : 0.f; }

// Compiler scheduling fence: memory ops cannot cross (keeps the load clause
// batched ahead of the WMMA burst; emits no instruction).
__device__ __forceinline__ void sched_fence() { asm volatile("" ::: "memory"); }

// WGP-scope prefetch: pull line into all cache levels (stop on WGP$ hit).
__device__ __forceinline__ void prefetch_wgp(const void* p) {
    asm volatile("global_prefetch_b8 %0, off" :: "v"(p));
}

// ---------------------------------------------------------------------------
// Repack row-major f32 W[K x N] into bf16 WMMA B-fragments.
// Fragment (kt, nt) = 32 lanes x 16 bf16 contiguous (1 KB):
//   lane l: column n = nt*16 + (l&15), K = kt*32 + (l&16)/2 + (j&7) + 2*(j&8), j = 0..15
// ---------------------------------------------------------------------------
__global__ void pack_w_frag(const float* __restrict__ W, __bf16* __restrict__ WP,
                            int N, int NT, int total)
{
    int p = blockIdx.x * blockDim.x + threadIdx.x;
    if (p >= total) return;
    int j    = p & 15;
    int lane = (p >> 4) & 31;
    int frag = p >> 9;               // kt*NT + nt
    int nt   = frag % NT;
    int kt   = frag / NT;
    int k = kt * 32 + ((lane & 16) >> 1) + (j & 7) + ((j & 8) << 1);
    int n = nt * 16 + (lane & 15);
    WP[p] = (__bf16)W[(size_t)k * N + n];
}

// ---------------------------------------------------------------------------
// Projection GEMM: Y[M x 640] = X[M x 512] @ W1[kt0*32 .. +512, :]
// block = 256 threads = 8 waves; wave -> 16 rows x 64 cols (4 WMMA tiles)
// grid.x = M/128, grid.y = 640/64
// ---------------------------------------------------------------------------
__global__ void proj_wmma_kernel(const float* __restrict__ X,
                                 const __bf16* __restrict__ W1P,
                                 float* __restrict__ Y, int kt0)
{
    const int lane    = threadIdx.x & 31;
    const int w       = threadIdx.x >> 5;
    const int rowTile = blockIdx.x * 8 + w;
    const int hi      = lane >> 4;       // 0: K 0..7/16..23  1: K 8..15/24..31
    const int ln      = lane & 15;
    const float* xrow = X + (size_t)(rowTile * 16 + ln) * E_DIM;

    v8f acc[4] = {};
#pragma unroll 1
    for (int kr = 0; kr < 16; ++kr) {
        const int kb = kr * 32 + hi * 8;
        const int kt = kt0 + kr;
        const __bf16* bpBase =
            W1P + ((size_t)((kt * 40 + blockIdx.y * 4) * 32 + lane)) * 16;

        // ---- issue ALL loads for this k-step first (one clause) ----
        float av[16];
        *(float4*)(&av[0])  = *(const float4*)(xrow + kb);
        *(float4*)(&av[4])  = *(const float4*)(xrow + kb + 4);
        *(float4*)(&av[8])  = *(const float4*)(xrow + kb + 16);
        *(float4*)(&av[12]) = *(const float4*)(xrow + kb + 20);
        BFragU bu[4];
#pragma unroll
        for (int j = 0; j < 4; ++j) {
            bu[j].q[0] = *(const uint4*)(bpBase + j * 512);
            bu[j].q[1] = *(const uint4*)(bpBase + j * 512 + 8);
        }
        // ---- WGP-scope prefetch of next k-step while WMMAs run ----
        if (kr + 1 < 16) {
            prefetch_wgp(xrow + kb + 32);
#pragma unroll
            for (int j = 0; j < 4; ++j)
                prefetch_wgp(bpBase + 40 * 512 + j * 512);
        }
        sched_fence();   // keep the load clause ahead of the WMMA burst

        v16bf a;
#pragma unroll
        for (int i = 0; i < 16; ++i) a[i] = (__bf16)av[i];

#pragma unroll
        for (int j = 0; j < 4; ++j)
            acc[j] = __builtin_amdgcn_wmma_f32_16x16x32_bf16(
                false, a, false, bu[j].v, (short)0, acc[j], false, false);
    }
    // C/D layout: VGPR r, lanes 0-15 -> M=r, lanes 16-31 -> M=8+r
#pragma unroll
    for (int j = 0; j < 4; ++j) {
        int col = (blockIdx.y * 4 + j) * 16 + ln;
        float* yp = Y + (size_t)(rowTile * 16 + hi * 8) * HID + col;
#pragma unroll
        for (int r = 0; r < 8; ++r) yp[(size_t)r * HID] = acc[j][r];
    }
}

// ---------------------------------------------------------------------------
// Fused joint kernel. Block = (b,t) x quarter of OUT; 256 threads = 8 waves.
// Stage 1: h[64 x 640] = relu(enc_proj[bt] + pred_proj[b,u] + b1) -> LDS bf16
// Stage 2: wave (wm,wn) computes rows u in [wm*16, +16), 128-col slice.
// ---------------------------------------------------------------------------
__global__ void joint_wmma_kernel(const float* __restrict__ enc_proj,
                                  const float* __restrict__ pred_proj,
                                  const float* __restrict__ b1,
                                  const __bf16* __restrict__ W2P,
                                  const float* __restrict__ b2,
                                  float* __restrict__ out)
{
    extern __shared__ char hsm[];                  // 64 * 1296 = 82944 B
    const int bt = blockIdx.x;                     // b*T + t
    const int nb = blockIdx.y;                     // 0..3 (256-col slab)
    const int b  = bt >> 8;                        // T = 256
    const float* encRow = enc_proj + (size_t)bt * HID;
    const float* predB  = pred_proj + (size_t)b * U_SZ * HID;

    // ---- build h tile in LDS (bf16, padded rows) ----
    for (int q = threadIdx.x; q < U_SZ * (HID / 4); q += 256) {
        int u  = q / (HID / 4);
        int k4 = (q - u * (HID / 4)) * 4;
        float4 p  = *(const float4*)(predB + (size_t)u * HID + k4);
        float4 e  = *(const float4*)(encRow + k4);
        float4 bb = *(const float4*)(b1 + k4);
        PackU pk;
        pk.b[0] = (__bf16)reluf(p.x + e.x + bb.x);
        pk.b[1] = (__bf16)reluf(p.y + e.y + bb.y);
        pk.b[2] = (__bf16)reluf(p.z + e.z + bb.z);
        pk.b[3] = (__bf16)reluf(p.w + e.w + bb.w);
        *(uint2*)(hsm + u * HROW_B + k4 * 2) = pk.u;
    }
    __syncthreads();

    const int lane = threadIdx.x & 31;
    const int w    = threadIdx.x >> 5;
    const int wm   = w & 3;        // u tile (16 rows)
    const int wn   = w >> 2;       // 0..1 (128-col half)
    const int hi   = lane >> 4;
    const int ln   = lane & 15;
    const char* hrow   = hsm + (wm * 16 + ln) * HROW_B;
    const int   ntBase = nb * 16 + wn * 8;   // n-tile index (of 64)

    v8f acc[8] = {};
#pragma unroll 1
    for (int kt = 0; kt < 20; ++kt) {
        // ---- issue ALL loads for this k-step first: 2 ds_read_b128 + 16 b128 ----
        const char* ha = hrow + (kt * 32 + hi * 8) * 2;
        BFragU au;
        au.q[0] = *(const uint4*)(ha);          // K base..base+7
        au.q[1] = *(const uint4*)(ha + 32);     // K base+16..+23

        const __bf16* bpBase =
            W2P + ((size_t)((kt * 64 + ntBase) * 32 + lane)) * 16;
        BFragU bu[8];
#pragma unroll
        for (int j = 0; j < 8; ++j) {
            bu[j].q[0] = *(const uint4*)(bpBase + j * 512);
            bu[j].q[1] = *(const uint4*)(bpBase + j * 512 + 8);
        }
        // ---- WGP-scope prefetch of next k-step's fragments ----
        if (kt + 1 < 20) {
            const __bf16* bpNext = bpBase + (size_t)64 * 512;
#pragma unroll
            for (int j = 0; j < 8; ++j)
                prefetch_wgp(bpNext + j * 512);
        }
        sched_fence();   // keep the 18-load clause ahead of the WMMA burst

        // ---- 8 back-to-back WMMAs (progressive loadcnt waits) ----
#pragma unroll
        for (int j = 0; j < 8; ++j)
            acc[j] = __builtin_amdgcn_wmma_f32_16x16x32_bf16(
                false, au.v, false, bu[j].v, (short)0, acc[j], false, false);
    }

    const size_t rowBase = (size_t)bt * U_SZ + wm * 16 + hi * 8;
#pragma unroll
    for (int j = 0; j < 8; ++j) {
        int col = (ntBase + j) * 16 + ln;
        float bias = b2[col];
        float* op = out + rowBase * OUT_D + col;
#pragma unroll
        for (int r = 0; r < 8; ++r) op[(size_t)r * OUT_D] = acc[j][r] + bias;
    }
}

// ---------------------------------------------------------------------------
extern "C" void kernel_launch(void* const* d_in, const int* in_sizes, int n_in,
                              void* d_out, int out_size, void* d_ws, size_t ws_size,
                              hipStream_t stream)
{
    const float* src_enc = (const float*)d_in[0];   // [8,256,512]
    const int*   src_len = (const int*)d_in[1];     // [8]
    const float* tgt_enc = (const float*)d_in[2];   // [8,64,512]
    const int*   tgt_len = (const int*)d_in[3];     // [8]
    const float* W1 = (const float*)d_in[4];        // [1024,640]
    const float* b1 = (const float*)d_in[5];        // [640]
    const float* W2 = (const float*)d_in[6];        // [640,1024]
    const float* b2 = (const float*)d_in[7];        // [1024]
    float* out = (float*)d_out;

    char* ws = (char*)d_ws;
    __bf16* W1P      = (__bf16*)(ws);               // 1,310,720 B
    __bf16* W2P      = (__bf16*)(ws + 1310720);     // 1,310,720 B
    float*  enc_proj = (float*)(ws + 2621440);      // 5,242,880 B
    float*  pred_proj= (float*)(ws + 7864320);      // 1,310,720 B

    {   // pack W1 [1024 x 640] -> fragments (NT = 640/16 = 40)
        int total = 1024 * HID;
        pack_w_frag<<<(total + 255) / 256, 256, 0, stream>>>(W1, W1P, HID, 40, total);
    }
    {   // pack W2 [640 x 1024] -> fragments (NT = 1024/16 = 64)
        int total = HID * OUT_D;
        pack_w_frag<<<(total + 255) / 256, 256, 0, stream>>>(W2, W2P, OUT_D, 64, total);
    }
    // enc_proj: M = B*T = 2048 rows, uses W1 rows [0,512)  -> kt0 = 0
    proj_wmma_kernel<<<dim3(2048 / 128, HID / 64), 256, 0, stream>>>(
        src_enc, W1P, enc_proj, 0);
    // pred_proj: M = B*U = 512 rows, uses W1 rows [512,1024) -> kt0 = 16
    proj_wmma_kernel<<<dim3(512 / 128, HID / 64), 256, 0, stream>>>(
        tgt_enc, W1P, pred_proj, 16);

    // fused joint: grid (B*T, OUT/256), 82944 B dynamic LDS
    joint_wmma_kernel<<<dim3(B_SZ * T_SZ, OUT_D / 256), 256, 82944, stream>>>(
        enc_proj, pred_proj, b1, W2P, b2, out);

    // tuple tail: (source_lengths, target_lengths) appended after joint_output
    const size_t joint_elems = (size_t)B_SZ * T_SZ * U_SZ * OUT_D;
    if ((size_t)out_size >= joint_elems + 16) {
        hipMemcpyAsync((char*)d_out + joint_elems * 4, src_len, 8 * sizeof(int),
                       hipMemcpyDeviceToDevice, stream);
        hipMemcpyAsync((char*)d_out + joint_elems * 4 + 32, tgt_len, 8 * sizeof(int),
                       hipMemcpyDeviceToDevice, stream);
    }
}
